// just_cosines_31258771980987
// MI455X (gfx1250) — matrix operand order repaired and checked by reading, not compiled, not run on hardware
//
#include <hip/hip_runtime.h>
#include <hip/hip_bf16.h>

typedef __attribute__((ext_vector_type(16))) __bf16 v16bf;
typedef __attribute__((ext_vector_type(8)))  __bf16 v8bf;
typedef __attribute__((ext_vector_type(8)))  float  v8f;

typedef unsigned int tdm_g0_t __attribute__((ext_vector_type(4)));  // uint32x4
typedef int          tdm_g1_t __attribute__((ext_vector_type(8)));  // int32x8
typedef int          tdm_g2_t __attribute__((ext_vector_type(4)));  // int32x4

#define N_ROWS   8192
#define DIM      768
#define KCH      (DIM / 32)        // 24 K-chunks for 16x16x32 WMMA
#define EPSN     1e-8f
#define RSTRIDE  396               // LDS row stride in dwords: 384 data + 3 pads * 4 dw
#define TILES_PER_SLICE 64
#define LOOKAHEAD 3                // B-fragment pipeline depth

#define HAS_TDM (__has_builtin(__builtin_amdgcn_tensor_load_to_lds) && \
                 __has_builtin(__builtin_amdgcn_s_wait_tensorcnt))
#define HAS_SGB __has_builtin(__builtin_amdgcn_sched_group_barrier)

// ---------------------------------------------------------------------------
// init: out = -inf  (cosines can be negative; harness poisons d_out)
// ---------------------------------------------------------------------------
__global__ __launch_bounds__(256) void jc_init(float* __restrict__ out)
{
    int i = blockIdx.x * 256 + threadIdx.x;
    if (i < N_ROWS) out[i] = -__builtin_huge_valf();
}

// ---------------------------------------------------------------------------
// Pass 1: row-normalize ex and ey, emit bf16 into workspace (one wave / row)
// ---------------------------------------------------------------------------
__global__ __launch_bounds__(256) void jc_normalize(
    const float* __restrict__ ex, const float* __restrict__ ey,
    __bf16* __restrict__ exn, __bf16* __restrict__ eyn)
{
    const int wave = threadIdx.x >> 5;
    const int lane = threadIdx.x & 31;
    const int row  = blockIdx.x * 8 + wave;        // 0 .. 16383

    const float* __restrict__ src;
    __bf16* __restrict__ dst;
    if (row < N_ROWS) {
        src = ex + (size_t)row * DIM;
        dst = exn + (size_t)row * DIM;
    } else {
        const int r = row - N_ROWS;
        src = ey + (size_t)r * DIM;
        dst = eyn + (size_t)r * DIM;
    }

    float s = 0.0f;
#pragma unroll
    for (int k = lane; k < DIM; k += 32) {
        float v = src[k];
        s += v * v;
    }
#pragma unroll
    for (int m = 16; m; m >>= 1) s += __shfl_xor(s, m, 32);

    const float scale = 1.0f / fmaxf(sqrtf(s), EPSN);
#pragma unroll
    for (int k = lane; k < DIM; k += 32) dst[k] = (__bf16)(src[k] * scale);
}

// ---------------------------------------------------------------------------
// CAS-based float atomic max (guaranteed to compile; only 64K of these total)
// ---------------------------------------------------------------------------
__device__ inline void atomicMaxF(float* a, float v)
{
    unsigned* ua  = (unsigned*)a;
    unsigned  old = __hip_atomic_load(ua, __ATOMIC_RELAXED, __HIP_MEMORY_SCOPE_AGENT);
    while (__uint_as_float(old) < v) {
        unsigned assumed = old;
        old = atomicCAS(ua, assumed, __float_as_uint(v));
        if (old == assumed) break;
    }
}

// ---------------------------------------------------------------------------
// TDM stage: DMA one eyn tile (16 rows x 768 bf16) into LDS with padding:
// pad 4 dwords after every 128 data dwords -> LDS row stride 396 dwords,
// giving a 12-bank skew per row (16 lanes -> 16 distinct banks).
// D# per CDNA5 ISA  (2D tensor, data_size=2B).
// ---------------------------------------------------------------------------
#if HAS_TDM
__device__ inline void tdm_stage(const __bf16* gsrc, unsigned lds_off)
{
    unsigned long long ga = (unsigned long long)(size_t)gsrc;
    tdm_g0_t g0;
    g0[0] = 1u;                                        // count=1, no gather
    g0[1] = lds_off;                                   // lds_addr [63:32]
    g0[2] = (unsigned)(ga & 0xffffffffu);              // global_addr lo
    g0[3] = (unsigned)((ga >> 32) & 0x1ffffffu)        // global_addr [56:32]
          | (2u << 30);                                // type=2 ("image")
    tdm_g1_t g1;
    g1[0] = (1 << 16)      // data_size = 2 bytes
          | (1 << 20)      // pad_enable
          | (6 << 22)      // pad_interval: 128 dwords
          | (3 << 25);     // pad_amount:   4 dwords (keeps 16B alignment)
    g1[1] = (int)(768u << 16);      // tensor_dim0 = 768      (bits 79:48)
    g1[2] = (int)(8192u << 16);     // tensor_dim1 = 8192     (bits 111:80)
    g1[3] = (int)(768u << 16);      // tile_dim0 = 768        (bits 127:112)
    g1[4] = 16;                     // tile_dim1 = 16, tile_dim2 = 0
    g1[5] = 768;                    // tensor_dim0_stride = 768 elements
    g1[6] = 0;
    g1[7] = 0;
    tdm_g2_t z4 = {0, 0, 0, 0};
#if defined(__clang_major__) && (__clang_major__ >= 23)
    tdm_g1_t z8 = {0, 0, 0, 0, 0, 0, 0, 0};
    __builtin_amdgcn_tensor_load_to_lds(g0, g1, z4, z4, z8, 0);
#else
    __builtin_amdgcn_tensor_load_to_lds(g0, g1, z4, z4, 0);
#endif
}
#endif

// B-fragment fetch from the padded LDS tile.
// B 32x16 k order used here (contiguous, same as A): lane = ey row (column N); lanes 0-15 hold K 0..15,
// lanes 16-31 hold K 16..31 -> 32B per lane per fragment (2 x ds_load_b128).
__device__ __forceinline__ v16bf load_bfrag(const unsigned* bbase, int kc, int kh)
{
    const int off = kc * 16 + kh * 8 + ((kc >> 3) << 2);   // + pad dwords
    v8bf lo = *(const v8bf*)(bbase + off);
    v8bf hi = *(const v8bf*)(bbase + off + 4);
    return __builtin_shufflevector(lo, hi,
            0,1,2,3,4,5,6,7,8,9,10,11,12,13,14,15);
}

// ---------------------------------------------------------------------------
// Pass 2: block = 8 waves = 128 exn rows, one ey slice of 64 tiles.
// Each wave: 16-row A tile register-resident (24 fragments), B fragments
// pipelined 3-deep from the TDM-staged LDS tile; schedule pinned with
// sched_group_barrier so the scheduler cannot re-serialize the loads.
// ---------------------------------------------------------------------------
__global__ __launch_bounds__(256) void jc_chamfer_max(
    const __bf16* __restrict__ exn, const __bf16* __restrict__ eyn,
    float* __restrict__ out)
{
    __shared__ __align__(16) unsigned ldsB[2][16 * RSTRIDE];   // ~49.5 KB

    const int wave    = threadIdx.x >> 5;
    const int lane    = threadIdx.x & 31;
    const int l15     = lane & 15;
    const int kh      = lane >> 4;                  // 16-lane half
    const int exBlock = blockIdx.x >> 3;            // 0..63  (128 rows each)
    const int slice   = blockIdx.x & 7;             // 0..7   (64 ey tiles each)
    const int rowBase = exBlock * 128 + wave * 16;  // this wave's 16 ex rows

    // --- A tile: lane holds ex row (rowBase + l15).
    // A 16x32 k order used here (contiguous, same as B): lanes 0-15 hold K 0..15,
    // lanes 16-31 hold K 16..31  (two 16B loads per fragment).
    const __bf16* __restrict__ arow = exn + (size_t)(rowBase + l15) * DIM;
    v16bf afr[KCH];
#pragma unroll
    for (int kc = 0; kc < KCH; ++kc) {
        v8bf lo = *(const v8bf*)(arow + kc * 32 + 16 * kh);
        v8bf hi = *(const v8bf*)(arow + kc * 32 + 16 * kh + 8);
        afr[kc] = __builtin_shufflevector(lo, hi,
                   0,1,2,3,4,5,6,7,8,9,10,11,12,13,14,15);
    }

    const float NEG = -__builtin_huge_valf();
    v8f vmax = {NEG, NEG, NEG, NEG, NEG, NEG, NEG, NEG};

    const unsigned lds0 = (unsigned)(size_t)(void*)&ldsB[0][0];
    const unsigned lds1 = (unsigned)(size_t)(void*)&ldsB[1][0];

#if HAS_TDM
    if (wave == 0)
        tdm_stage(eyn + (size_t)(slice * TILES_PER_SLICE) * 16 * DIM, lds0);
#endif

#pragma unroll 1
    for (int tt = 0; tt < TILES_PER_SLICE; ++tt) {
#if HAS_TDM
        if (wave == 0) {
            if (tt + 1 < TILES_PER_SLICE) {
                const int tile = slice * TILES_PER_SLICE + tt + 1;
                tdm_stage(eyn + (size_t)tile * 16 * DIM,
                          ((tt + 1) & 1) ? lds1 : lds0);
                __builtin_amdgcn_s_wait_tensorcnt(1);  // tile tt arrived
            } else {
                __builtin_amdgcn_s_wait_tensorcnt(0);
            }
        }
#else
        // fallback: cooperative copy of tile tt into the current buffer
        {
            const int tile = slice * TILES_PER_SLICE + tt;
            unsigned* dstb = &ldsB[tt & 1][0];
            const unsigned* srcb =
                (const unsigned*)(eyn + (size_t)tile * 16 * DIM);
            for (int i = threadIdx.x; i < 16 * 384; i += 256) {
                int r = i / 384, d = i % 384;
                dstb[r * RSTRIDE + d + ((d >> 7) << 2)] = srcb[r * 384 + d];
            }
        }
#endif
        __syncthreads();   // tile tt visible to all waves

        // --- 16x16x768 tile GEMM from LDS, B pipelined LOOKAHEAD deep.
        const unsigned* bbase = &ldsB[tt & 1][(unsigned)l15 * RSTRIDE];
        v8f acc = {0.f, 0.f, 0.f, 0.f, 0.f, 0.f, 0.f, 0.f};

        v16bf b0 = load_bfrag(bbase, 0, kh);
        v16bf b1 = load_bfrag(bbase, 1, kh);
        v16bf b2 = load_bfrag(bbase, 2, kh);
#pragma unroll
        for (int kc = 0; kc < KCH - LOOKAHEAD; ++kc) {
            v16bf p = load_bfrag(bbase, kc + LOOKAHEAD, kh);
            acc = __builtin_amdgcn_wmma_f32_16x16x32_bf16(
                      false, afr[kc], false, b0, (short)0, acc, false, false);
            b0 = b1; b1 = b2; b2 = p;
        }
        acc = __builtin_amdgcn_wmma_f32_16x16x32_bf16(
                  false, afr[KCH - 3], false, b0, (short)0, acc, false, false);
        acc = __builtin_amdgcn_wmma_f32_16x16x32_bf16(
                  false, afr[KCH - 2], false, b1, (short)0, acc, false, false);
        acc = __builtin_amdgcn_wmma_f32_16x16x32_bf16(
                  false, afr[KCH - 1], false, b2, (short)0, acc, false, false);

#if HAS_SGB
        // Pin the schedule of this region: DS-reads run 3 fragments ahead
        // of the consuming WMMA.  0x100 = DS-read group, 0x008 = WMMA group.
        __builtin_amdgcn_sched_group_barrier(0x100, 2 * LOOKAHEAD, 0);
#pragma unroll
        for (int kc = 0; kc < KCH - LOOKAHEAD; ++kc) {
            __builtin_amdgcn_sched_group_barrier(0x008, 1, 0);
            __builtin_amdgcn_sched_group_barrier(0x100, 2, 0);
        }
        __builtin_amdgcn_sched_group_barrier(0x008, LOOKAHEAD, 0);
#endif

#pragma unroll
        for (int r = 0; r < 8; ++r) vmax[r] = fmaxf(vmax[r], acc[r]);

        __syncthreads();   // all reads of this buffer done before next DMA
    }

    // --- reduce across the 16 lanes sharing the same M rows.
    // D layout: VGPR r, lane l -> M = r + 8*(l>>4), N = l&15.
#pragma unroll
    for (int r = 0; r < 8; ++r) {
        float v = vmax[r];
        v = fmaxf(v, __shfl_xor(v, 1, 32));
        v = fmaxf(v, __shfl_xor(v, 2, 32));
        v = fmaxf(v, __shfl_xor(v, 4, 32));
        v = fmaxf(v, __shfl_xor(v, 8, 32));
        vmax[r] = v;
    }
    if (l15 == 0) {
#pragma unroll
        for (int r = 0; r < 8; ++r)
            atomicMaxF(&out[rowBase + kh * 8 + r], vmax[r]);
    }
}

// ---------------------------------------------------------------------------
extern "C" void kernel_launch(void* const* d_in, const int* in_sizes, int n_in,
                              void* d_out, int out_size, void* d_ws, size_t ws_size,
                              hipStream_t stream)
{
    const float* ex = (const float*)d_in[0];
    const float* ey = (const float*)d_in[1];
    float* out = (float*)d_out;

    // workspace: exn (8192x768 bf16) then eyn (8192x768 bf16) ~ 24 MB
    __bf16* exn = (__bf16*)d_ws;
    __bf16* eyn = exn + (size_t)N_ROWS * DIM;

    jc_init<<<N_ROWS / 256, 256, 0, stream>>>(out);
    jc_normalize<<<(2 * N_ROWS) / 8, 256, 0, stream>>>(ex, ey, exn, eyn);
    jc_chamfer_max<<<N_ROWS / 16, 256, 0, stream>>>(exn, eyn, out);
}
